// TransformerEncoderLayerQaN_64879775973735
// MI455X (gfx1250) — matrix operand order repaired and checked by reading, not compiled, not run on hardware
//
#include <hip/hip_runtime.h>

// ---------------------------------------------------------------------------
// Problem constants (from reference)
// ---------------------------------------------------------------------------
#define T_LEN   512
#define B_SZ    4
#define DM      1024          // d_model
#define DFF_    4096
#define NQ_     10
#define WIN_    64
#define NWIN    8             // T_LEN / WIN
#define ROWS    2048          // T_LEN * B_SZ
#define LKEY    192           // 3 * WIN (look_bwd=look_fwd=1)

typedef __bf16 bf16;
typedef __attribute__((ext_vector_type(8)))  bf16  v8bf;
typedef __attribute__((ext_vector_type(16))) bf16  v16bf;
typedef __attribute__((ext_vector_type(8)))  float v8f;

// ---------------------------------------------------------------------------
// fp32 <-> bf16 (round-to-nearest-even), plus hi/lo split helpers
// ---------------------------------------------------------------------------
static __device__ __forceinline__ bf16 f32_to_bf16(float f) {
  union { float f; unsigned u; } in; in.f = f;
  unsigned u = in.u;
  unsigned r = u + 0x7FFFu + ((u >> 16) & 1u);
  union { unsigned short s; bf16 b; } out; out.s = (unsigned short)(r >> 16);
  return out.b;
}
static __device__ __forceinline__ float bf16_to_f32(bf16 b) {
  union { unsigned short s; bf16 b; } in; in.b = b;
  union { unsigned u; float f; } out; out.u = ((unsigned)in.s) << 16;
  return out.f;
}

// ---------------------------------------------------------------------------
// WMMA fragment loaders (CDNA5 16x16x32 bf16, wave32).
// A (16xK, lane = M row):  lanes 0-15 hold K [k0..k0+7] then [k0+16..k0+23],
//                          lanes 16-31 hold K [k0+8..k0+15] then [k0+24..k0+31].
// B (Kx16, lane = N col):  same K layout as A: lanes 0-15 hold K [k0..k0+7] then [k0+16..k0+23],
//                          lanes 16-31 hold K [k0+8..k0+15] then [k0+24..k0+31].
// Both our operands are row-major with K contiguous (NT GEMM), so these are
// plain 16B/32B vector loads.
// ---------------------------------------------------------------------------
static __device__ __forceinline__ v16bf load_a_frag(const bf16* row, int lane, int k0) {
  const int kb = k0 + ((lane & 16) ? 8 : 0);
  union { v16bf v; v8bf h[2]; } f;
  f.h[0] = *(const v8bf*)(row + kb);
  f.h[1] = *(const v8bf*)(row + kb + 16);
  return f.v;
}
static __device__ __forceinline__ v16bf load_b_frag(const bf16* row, int lane, int k0) {
  const int kb = k0 + ((lane & 16) ? 8 : 0);
  union { v16bf v; v8bf h[2]; } f;
  f.h[0] = *(const v8bf*)(row + kb);
  f.h[1] = *(const v8bf*)(row + kb + 16);
  return f.v;
}

// ---------------------------------------------------------------------------
// Kernel 1: per-head query normalization.  qn[n,d] = q/(||q_head||+1e-6)/8
// ---------------------------------------------------------------------------
__global__ __launch_bounds__(1024)
void qn_kernel(const float* __restrict__ queries, float* __restrict__ qn) {
  __shared__ float s[DM];
  const int n = blockIdx.x, d = threadIdx.x;
  const float qv = queries[(size_t)n * DM + d];
  s[d] = qv * qv;
  __syncthreads();
  const int h0 = d & ~63;             // head of 64 dims
  float ss = 0.f;
  #pragma unroll 8
  for (int i = 0; i < 64; ++i) ss += s[h0 + i];
  qn[(size_t)n * DM + d] = qv / ((sqrtf(ss) + 1e-6f) * 8.0f);
}

// ---------------------------------------------------------------------------
// Kernel 2: split an fp32 array into bf16 hi/lo pair (for split-bf16 WMMA).
// ---------------------------------------------------------------------------
__global__ __launch_bounds__(256)
void split_kernel(const float* __restrict__ src, bf16* __restrict__ hi,
                  bf16* __restrict__ lo, int n) {
  int i = blockIdx.x * 256 + threadIdx.x;
  const int stride = gridDim.x * 256;
  for (; i < n; i += stride) {
    const float f = src[i];
    const bf16 h = f32_to_bf16(f);
    hi[i] = h;
    lo[i] = f32_to_bf16(f - bf16_to_f32(h));
  }
}

// ---------------------------------------------------------------------------
// Kernel 3: qa_block.  Since q is time-invariant, attention output is one
// vector per (b, window).  One block per (b,w); loops over the NQ queries,
// builds effective key weights e_j = sum_n wk[n]*softmax_n(j), then does a
// single weighted sum over the 192 window keys.
// ---------------------------------------------------------------------------
__global__ __launch_bounds__(256)
void qa_kernel(const float* __restrict__ src, const float* __restrict__ qn,
               const float* __restrict__ wk, float* __restrict__ qa) {
  __shared__ float s_sc[LKEY];
  __shared__ float s_e[LKEY];
  __shared__ float s_red[256];
  const int tid  = threadIdx.x;
  const int lane = tid & 31;
  const int wv   = tid >> 5;
  const int b    = blockIdx.x >> 3;
  const int w    = blockIdx.x & 7;
  const int kt0  = (w - 1) * WIN_;
  const float scale = 0.03125f;                 // 1024^-0.5
  if (tid < LKEY) s_e[tid] = 0.f;
  __syncthreads();

  for (int n = 0; n < NQ_; ++n) {
    const float* q = qn + (size_t)n * DM;
    // scores: one wave per key, lane-strided dot of length 1024
    for (int j = wv; j < LKEY; j += 8) {
      const int kt = kt0 + j;
      float p = 0.f;
      if (kt >= 0 && kt < T_LEN) {
        const float* kv = src + ((size_t)kt * B_SZ + b) * DM;
        for (int kk = lane; kk < DM; kk += 32) p += q[kk] * kv[kk];
      }
      #pragma unroll
      for (int o = 16; o > 0; o >>= 1) p += __shfl_xor(p, o, 32);
      if (lane == 0)
        s_sc[j] = (kt >= 0 && kt < T_LEN) ? p * scale : -3.402823466e38f;
    }
    __syncthreads();
    // softmax over 192 (invalid keys carry -FLT_MAX like the reference)
    float v = (tid < LKEY) ? s_sc[tid] : -3.402823466e38f;
    s_red[tid] = v; __syncthreads();
    for (int o = 128; o > 0; o >>= 1) {
      if (tid < o) s_red[tid] = fmaxf(s_red[tid], s_red[tid + o]);
      __syncthreads();
    }
    const float m = s_red[0]; __syncthreads();
    const float e = (tid < LKEY) ? expf(v - m) : 0.f;
    s_red[tid] = e; __syncthreads();
    for (int o = 128; o > 0; o >>= 1) {
      if (tid < o) s_red[tid] += s_red[tid + o];
      __syncthreads();
    }
    const float S = s_red[0]; __syncthreads();
    if (tid < LKEY) s_e[tid] += wk[n] * (e / S);
    __syncthreads();
  }

  // weighted sum of values with the combined weights
  float a0 = 0.f, a1 = 0.f, a2 = 0.f, a3 = 0.f;
  for (int j = 0; j < LKEY; ++j) {
    const int kt = kt0 + j;
    if (kt < 0 || kt >= T_LEN) continue;
    const float ej = s_e[j];
    const float* kv = src + ((size_t)kt * B_SZ + b) * DM;
    a0 += ej * kv[tid];
    a1 += ej * kv[tid + 256];
    a2 += ej * kv[tid + 512];
    a3 += ej * kv[tid + 768];
  }
  float* o = qa + (size_t)(b * NWIN + w) * DM;
  o[tid] = a0; o[tid + 256] = a1; o[tid + 512] = a2; o[tid + 768] = a3;
}

// ---------------------------------------------------------------------------
// Kernel 4: residual + LayerNorm1, also emitting the bf16 hi/lo split of x
// (GEMM1's A operand).  One block per row (t*B+b).
// ---------------------------------------------------------------------------
__global__ __launch_bounds__(256)
void ln1_kernel(const float* __restrict__ src, const float* __restrict__ qa,
                const float* __restrict__ g, const float* __restrict__ be,
                float* __restrict__ x, bf16* __restrict__ xh, bf16* __restrict__ xl) {
  __shared__ float s_red[256];
  const int r = blockIdx.x, tid = threadIdx.x;
  const int t = r / B_SZ, b = r % B_SZ;
  const float* qrow = qa + (size_t)(b * NWIN + (t >> 6)) * DM;
  const float* srow = src + (size_t)r * DM;
  float v[4]; float s = 0.f;
  #pragma unroll
  for (int q = 0; q < 4; ++q) { const int d = tid + q * 256; v[q] = srow[d] + qrow[d]; s += v[q]; }
  s_red[tid] = s; __syncthreads();
  for (int o = 128; o > 0; o >>= 1) { if (tid < o) s_red[tid] += s_red[tid + o]; __syncthreads(); }
  const float mu = s_red[0] * (1.f / DM); __syncthreads();
  float s2 = 0.f;
  #pragma unroll
  for (int q = 0; q < 4; ++q) { const float d = v[q] - mu; s2 += d * d; }
  s_red[tid] = s2; __syncthreads();
  for (int o = 128; o > 0; o >>= 1) { if (tid < o) s_red[tid] += s_red[tid + o]; __syncthreads(); }
  const float rstd = rsqrtf(s_red[0] * (1.f / DM) + 1e-5f);
  #pragma unroll
  for (int q = 0; q < 4; ++q) {
    const int d = tid + q * 256;
    const float xv = (v[q] - mu) * rstd * g[d] + be[d];
    const size_t idx = (size_t)r * DM + d;
    x[idx] = xv;
    const bf16 h = f32_to_bf16(xv);
    xh[idx] = h;
    xl[idx] = f32_to_bf16(xv - bf16_to_f32(h));
  }
}

// ---------------------------------------------------------------------------
// Kernel 5: GEMM1  h = relu(x @ w1^T + b1), split-bf16 WMMA, NT layout.
// One wave computes a 16(M) x 64(N) tile with 4 accumulators sharing the A
// fragment; hi/lo bf16 of h is written directly (GEMM2's A operand).
// grid = (ROWS/16, DFF/(64*8)), block = 256 (8 waves).
// ---------------------------------------------------------------------------
__global__ __launch_bounds__(256)
void gemm1_kernel(const bf16* __restrict__ ahm, const bf16* __restrict__ alm,
                  const bf16* __restrict__ bhm, const bf16* __restrict__ blm,
                  const float* __restrict__ bias,
                  bf16* __restrict__ hh, bf16* __restrict__ hl) {
  const int lane = threadIdx.x & 31;
  const int wv   = threadIdx.x >> 5;
  const int m0   = blockIdx.x * 16;
  const int n0   = (blockIdx.y * 8 + wv) * 64;
  const int lr   = lane & 15;
  v8f acc[4] = {};
  const bf16* arh = ahm + (size_t)(m0 + lr) * DM;
  const bf16* arl = alm + (size_t)(m0 + lr) * DM;
  for (int k0 = 0; k0 < DM; k0 += 32) {
    const v16bf ah = load_a_frag(arh, lane, k0);
    const v16bf al = load_a_frag(arl, lane, k0);
    #pragma unroll
    for (int a = 0; a < 4; ++a) {
      const bf16* brh = bhm + (size_t)(n0 + a * 16 + lr) * DM;
      const bf16* brl = blm + (size_t)(n0 + a * 16 + lr) * DM;
      const v16bf bh = load_b_frag(brh, lane, k0);
      const v16bf bl = load_b_frag(brl, lane, k0);
      acc[a] = __builtin_amdgcn_wmma_f32_16x16x32_bf16(false, ah, false, bh, (short)0, acc[a], false, false);
      acc[a] = __builtin_amdgcn_wmma_f32_16x16x32_bf16(false, ah, false, bl, (short)0, acc[a], false, false);
      acc[a] = __builtin_amdgcn_wmma_f32_16x16x32_bf16(false, al, false, bh, (short)0, acc[a], false, false);
    }
  }
  const int mb = m0 + ((lane & 16) ? 8 : 0);   // C layout: M = vgpr + 8*(lane>=16)
  #pragma unroll
  for (int a = 0; a < 4; ++a) {
    const int n = n0 + a * 16 + lr;
    const float bv = bias[n];
    #pragma unroll
    for (int e = 0; e < 8; ++e) {
      float c = fmaxf(acc[a][e] + bv, 0.f);
      const size_t idx = (size_t)(mb + e) * DFF_ + n;
      const bf16 h = f32_to_bf16(c);
      hh[idx] = h;
      hl[idx] = f32_to_bf16(c - bf16_to_f32(h));
    }
  }
}

// ---------------------------------------------------------------------------
// Kernel 6: GEMM2  y = h @ w2^T + b2, fp32 out.  grid = (ROWS/16, DM/(64*8)).
// ---------------------------------------------------------------------------
__global__ __launch_bounds__(256)
void gemm2_kernel(const bf16* __restrict__ ahm, const bf16* __restrict__ alm,
                  const bf16* __restrict__ bhm, const bf16* __restrict__ blm,
                  const float* __restrict__ bias, float* __restrict__ y) {
  const int lane = threadIdx.x & 31;
  const int wv   = threadIdx.x >> 5;
  const int m0   = blockIdx.x * 16;
  const int n0   = (blockIdx.y * 8 + wv) * 64;
  const int lr   = lane & 15;
  v8f acc[4] = {};
  const bf16* arh = ahm + (size_t)(m0 + lr) * DFF_;
  const bf16* arl = alm + (size_t)(m0 + lr) * DFF_;
  for (int k0 = 0; k0 < DFF_; k0 += 32) {
    const v16bf ah = load_a_frag(arh, lane, k0);
    const v16bf al = load_a_frag(arl, lane, k0);
    #pragma unroll
    for (int a = 0; a < 4; ++a) {
      const bf16* brh = bhm + (size_t)(n0 + a * 16 + lr) * DFF_;
      const bf16* brl = blm + (size_t)(n0 + a * 16 + lr) * DFF_;
      const v16bf bh = load_b_frag(brh, lane, k0);
      const v16bf bl = load_b_frag(brl, lane, k0);
      acc[a] = __builtin_amdgcn_wmma_f32_16x16x32_bf16(false, ah, false, bh, (short)0, acc[a], false, false);
      acc[a] = __builtin_amdgcn_wmma_f32_16x16x32_bf16(false, ah, false, bl, (short)0, acc[a], false, false);
      acc[a] = __builtin_amdgcn_wmma_f32_16x16x32_bf16(false, al, false, bh, (short)0, acc[a], false, false);
    }
  }
  const int mb = m0 + ((lane & 16) ? 8 : 0);
  #pragma unroll
  for (int a = 0; a < 4; ++a) {
    const int n = n0 + a * 16 + lr;
    const float bv = bias[n];
    #pragma unroll
    for (int e = 0; e < 8; ++e)
      y[(size_t)(mb + e) * DM + n] = acc[a][e] + bv;
  }
}

// ---------------------------------------------------------------------------
// Kernel 7: residual + LayerNorm2 -> final output (src + (x - src) == x).
// ---------------------------------------------------------------------------
__global__ __launch_bounds__(256)
void ln2_kernel(const float* __restrict__ x, const float* __restrict__ y,
                const float* __restrict__ g, const float* __restrict__ be,
                float* __restrict__ out) {
  __shared__ float s_red[256];
  const int r = blockIdx.x, tid = threadIdx.x;
  const float* xr = x + (size_t)r * DM;
  const float* yr = y + (size_t)r * DM;
  float v[4]; float s = 0.f;
  #pragma unroll
  for (int q = 0; q < 4; ++q) { const int d = tid + q * 256; v[q] = xr[d] + yr[d]; s += v[q]; }
  s_red[tid] = s; __syncthreads();
  for (int o = 128; o > 0; o >>= 1) { if (tid < o) s_red[tid] += s_red[tid + o]; __syncthreads(); }
  const float mu = s_red[0] * (1.f / DM); __syncthreads();
  float s2 = 0.f;
  #pragma unroll
  for (int q = 0; q < 4; ++q) { const float d = v[q] - mu; s2 += d * d; }
  s_red[tid] = s2; __syncthreads();
  for (int o = 128; o > 0; o >>= 1) { if (tid < o) s_red[tid] += s_red[tid + o]; __syncthreads(); }
  const float rstd = rsqrtf(s_red[0] * (1.f / DM) + 1e-5f);
  #pragma unroll
  for (int q = 0; q < 4; ++q) {
    const int d = tid + q * 256;
    out[(size_t)r * DM + d] = (v[q] - mu) * rstd * g[d] + be[d];
  }
}

// ---------------------------------------------------------------------------
// Host-side launcher
// ---------------------------------------------------------------------------
extern "C" void kernel_launch(void* const* d_in, const int* in_sizes, int n_in,
                              void* d_out, int out_size, void* d_ws, size_t ws_size,
                              hipStream_t stream) {
  (void)in_sizes; (void)n_in; (void)out_size; (void)ws_size;
  const float* src     = (const float*)d_in[0];
  const float* queries = (const float*)d_in[1];
  const float* wk      = (const float*)d_in[2];
  const float* w1      = (const float*)d_in[3];
  const float* b1      = (const float*)d_in[4];
  const float* w2      = (const float*)d_in[5];
  const float* b2      = (const float*)d_in[6];
  const float* g1      = (const float*)d_in[7];
  const float* be1     = (const float*)d_in[8];
  const float* g2      = (const float*)d_in[9];
  const float* be2     = (const float*)d_in[10];
  float* out = (float*)d_out;

  char* base = (char*)d_ws;
  size_t off = 0;
  auto alloc = [&](size_t bytes) -> char* {
    char* r = base + off;
    off += (bytes + 255) & ~(size_t)255;
    return r;
  };
  float* qn  = (float*)alloc((size_t)NQ_ * DM * 4);
  float* qa  = (float*)alloc((size_t)B_SZ * NWIN * DM * 4);
  float* x   = (float*)alloc((size_t)ROWS * DM * 4);
  bf16*  xh  = (bf16*) alloc((size_t)ROWS * DM * 2);
  bf16*  xl  = (bf16*) alloc((size_t)ROWS * DM * 2);
  bf16*  w1h = (bf16*) alloc((size_t)DFF_ * DM * 2);
  bf16*  w1l = (bf16*) alloc((size_t)DFF_ * DM * 2);
  bf16*  w2h = (bf16*) alloc((size_t)DM * DFF_ * 2);
  bf16*  w2l = (bf16*) alloc((size_t)DM * DFF_ * 2);
  bf16*  hh  = (bf16*) alloc((size_t)ROWS * DFF_ * 2);
  bf16*  hl  = (bf16*) alloc((size_t)ROWS * DFF_ * 2);
  float* y   = (float*)alloc((size_t)ROWS * DM * 4);

  qn_kernel   <<<NQ_, 1024, 0, stream>>>(queries, qn);
  split_kernel<<<2048, 256, 0, stream>>>(w1, w1h, w1l, DFF_ * DM);
  split_kernel<<<2048, 256, 0, stream>>>(w2, w2h, w2l, DM * DFF_);
  qa_kernel   <<<B_SZ * NWIN, 256, 0, stream>>>(src, qn, wk, qa);
  ln1_kernel  <<<ROWS, 256, 0, stream>>>(src, qa, g1, be1, x, xh, xl);
  gemm1_kernel<<<dim3(ROWS / 16, DFF_ / 512), 256, 0, stream>>>(xh, xl, w1h, w1l, b1, hh, hl);
  gemm2_kernel<<<dim3(ROWS / 16, DM / 512),  256, 0, stream>>>(hh, hl, w2h, w2l, b2, y);
  ln2_kernel  <<<ROWS, 256, 0, stream>>>(x, y, g2, be2, out);
}
